// SelfAttention2d_14035953123469
// MI455X (gfx1250) — matrix operand order repaired and checked by reading, not compiled, not run on hardware
//
#include <hip/hip_runtime.h>
#include <hip/hip_bf16.h>

typedef _Float16 f16;
typedef __attribute__((ext_vector_type(16))) _Float16 v16h;
typedef __attribute__((ext_vector_type(8)))  _Float16 v8h;
typedef __attribute__((ext_vector_type(8)))  float    v8f;

// Build a 16x32 f16 A-fragment for one lane from two contiguous 8-f16 chunks.
// Per CDNA5 ISA: lanes 0-15 hold K=0..7 (V0-3) and K=16..23 (V4-7);
// lanes 16-31 hold K=8..15 and K=24..31.  Callers pass row+16*half / row+16*half+8 (contiguous k, as B).
__device__ __forceinline__ v16h make_a_frag(const f16* p_lo, const f16* p_hi) {
  v8h lo = *(const v8h*)p_lo;
  v8h hi = *(const v8h*)p_hi;
  v16h a;
#pragma unroll
  for (int i = 0; i < 8; ++i) { a[i] = lo[i]; a[i + 8] = hi[i]; }
  return a;
}

// ---------------- weights f32 -> f16 ----------------
__global__ void cvt_f16_kernel(const float* __restrict__ src, f16* __restrict__ dst, int n) {
  int i = blockIdx.x * blockDim.x + threadIdx.x;
  int stride = gridDim.x * blockDim.x;
  for (; i < n; i += stride) dst[i] = (f16)src[i];
}

// ---------------- GroupNorm -> xn_t [n][t][c] f16 ----------------
__global__ void groupnorm_kernel(const float* __restrict__ x,
                                 const float* __restrict__ gw,
                                 const float* __restrict__ gb,
                                 f16* __restrict__ xnt) {
  const int g = blockIdx.x;    // 8 groups of 32 channels
  const int n = blockIdx.y;    // batch
  const int tid = threadIdx.x; // 256 threads
  const float* xb = x + ((size_t)n * 256 + g * 32) * 1024;

  __shared__ float rs[256], rs2[256];
  float s = 0.f, s2 = 0.f;
  for (int i = tid; i < 32 * 1024; i += 256) {
    float v = xb[i];
    s += v; s2 += v * v;
  }
  rs[tid] = s; rs2[tid] = s2;
  __syncthreads();
  for (int off = 128; off > 0; off >>= 1) {
    if (tid < off) { rs[tid] += rs[tid + off]; rs2[tid] += rs2[tid + off]; }
    __syncthreads();
  }
  const float mean = rs[0] * (1.f / 32768.f);
  const float var  = rs2[0] * (1.f / 32768.f) - mean * mean;
  const float rstd = rsqrtf(var + 1e-5f);

  for (int i = tid; i < 32 * 1024; i += 256) {
    int cl = i >> 10;
    int t  = i & 1023;
    int c  = g * 32 + cl;
    float v = (xb[i] - mean) * rstd * gw[c] + gb[c];
    xnt[((size_t)n * 1024 + t) * 256 + c] = (f16)v;
  }
}

// ---------------- QKV GEMM: [768x256] x [256x1024] per batch ----------------
// Each wave: 64(M) x 32(N) tile; block = 8 waves -> 128x128; K stepped by 32.
__global__ void qkv_gemm_kernel(const f16* __restrict__ W,      // [768][256]
                                const float* __restrict__ bias, // [768]
                                const f16* __restrict__ xnt,    // [n][1024][256]
                                f16* __restrict__ qt,           // [n][32][1024][8] (pre-scaled)
                                f16* __restrict__ kt,           // [n][32][1024][8]
                                f16* __restrict__ vt)           // [n][32][8][1024]
{
  const int n    = blockIdx.z;
  const int lane = threadIdx.x & 31;
  const int w    = threadIdx.x >> 5;
  const int half = lane >> 4;
  const int ln   = lane & 15;
  const int m0   = blockIdx.x * 128 + (w & 1) * 64;
  const int n0   = blockIdx.y * 128 + (w >> 1) * 32;

  v8f acc[4][2] = {};
  const f16* xb = xnt + (size_t)n * 1024 * 256;

  for (int kk = 0; kk < 256; kk += 32) {
    v16h bfrag[2];
#pragma unroll
    for (int nt = 0; nt < 2; ++nt)
      bfrag[nt] = *(const v16h*)(xb + (size_t)(n0 + nt * 16 + ln) * 256 + kk + 16 * half);
#pragma unroll
    for (int mt = 0; mt < 4; ++mt) {
      const f16* rowp = W + (size_t)(m0 + mt * 16 + ln) * 256 + kk;
      v16h afrag = make_a_frag(rowp + 16 * half, rowp + 16 * half + 8);
#pragma unroll
      for (int nt = 0; nt < 2; ++nt)
        acc[mt][nt] = __builtin_amdgcn_wmma_f32_16x16x32_f16(
            false, afrag, false, bfrag[nt], (short)0, acc[mt][nt], false, false);
    }
  }

  const float qscale = 0.35355339059327373f; // 1/sqrt(8)
#pragma unroll
  for (int mt = 0; mt < 4; ++mt)
#pragma unroll
    for (int nt = 0; nt < 2; ++nt)
#pragma unroll
      for (int r = 0; r < 8; ++r) {
        int o = m0 + mt * 16 + r + 8 * half;   // D row = r + 8*half
        int t = n0 + nt * 16 + ln;             // D col = lane&15
        float v = acc[mt][nt][r] + bias[o];
        int sec  = o >> 8;
        int oo   = o & 255;
        int head = oo >> 3;
        int d    = oo & 7;
        if (sec == 0)      qt[(((size_t)n * 32 + head) * 1024 + t) * 8 + d] = (f16)(v * qscale);
        else if (sec == 1) kt[(((size_t)n * 32 + head) * 1024 + t) * 8 + d] = (f16)v;
        else               vt[(((size_t)n * 32 + head) * 8 + d) * 1024 + t] = (f16)v;
      }
}

// ---------------- Flash attention: one wave per 16-row s-tile ----------------
__global__ void attn_kernel(const f16* __restrict__ qt,
                            const f16* __restrict__ kt,
                            const f16* __restrict__ vt,
                            f16* __restrict__ yt)   // [n][1024][256]
{
  const int schunk = blockIdx.x;  // 8 chunks of 128 rows
  const int head   = blockIdx.y;  // 32 heads
  const int n      = blockIdx.z;  // batch
  const int lane = threadIdx.x & 31;
  const int w    = threadIdx.x >> 5;
  const int half = lane >> 4;
  const int ln   = lane & 15;
  const int s0   = schunk * 128 + w * 16;
  const size_t nh = (size_t)n * 32 + head;

  __shared__ f16 Plds[8][16][40];  // per-wave 16x32 P tile (padded, 16B-aligned rows)

  // q A-fragment: K = head_dim (8) zero-padded to 32.
  v16h qa = {};
  if (half == 0) {
    v8h qv = *(const v8h*)(qt + (nh * 1024 + s0 + ln) * 8);
#pragma unroll
    for (int i = 0; i < 8; ++i) qa[i] = qv[i];
  }

  v8f acc = {};
  float m_run[8], l_run[8];
#pragma unroll
  for (int r = 0; r < 8; ++r) { m_run[r] = -1e30f; l_run[r] = 0.f; }

  for (int tc = 0; tc < 32; ++tc) {
    const int t0 = tc * 32;
    // scores: 16(s) x 32(t) via two WMMAs, K padded
    v16h kb0 = {}, kb1 = {};
    if (half == 0) {
      v8h kv0 = *(const v8h*)(kt + (nh * 1024 + t0 + ln) * 8);
      v8h kv1 = *(const v8h*)(kt + (nh * 1024 + t0 + 16 + ln) * 8);
#pragma unroll
      for (int i = 0; i < 8; ++i) { kb0[i] = kv0[i]; kb1[i] = kv1[i]; }
    }
    v8f sc0 = {}, sc1 = {};
    sc0 = __builtin_amdgcn_wmma_f32_16x16x32_f16(false, qa, false, kb0, (short)0, sc0, false, false);
    sc1 = __builtin_amdgcn_wmma_f32_16x16x32_f16(false, qa, false, kb1, (short)0, sc1, false, false);

    // online softmax; D row = r + 8*half lives in the 16 lanes of this half
#pragma unroll
    for (int r = 0; r < 8; ++r) {
      float c = fmaxf(sc0[r], sc1[r]);
      c = fmaxf(c, __shfl_xor(c, 1, 32));
      c = fmaxf(c, __shfl_xor(c, 2, 32));
      c = fmaxf(c, __shfl_xor(c, 4, 32));
      c = fmaxf(c, __shfl_xor(c, 8, 32));
      float nm    = fmaxf(m_run[r], c);
      float scale = __expf(m_run[r] - nm);
      m_run[r] = nm;
      float e0 = __expf(sc0[r] - nm);
      float e1 = __expf(sc1[r] - nm);
      float rs = e0 + e1;
      rs += __shfl_xor(rs, 1, 32);
      rs += __shfl_xor(rs, 2, 32);
      rs += __shfl_xor(rs, 4, 32);
      rs += __shfl_xor(rs, 8, 32);
      l_run[r] = l_run[r] * scale + rs;
      acc[r]   = acc[r] * scale;
      Plds[w][r + 8 * half][ln]      = (f16)e0;
      Plds[w][r + 8 * half][ln + 16] = (f16)e1;
    }
    // same-wave DS store->load is in-order; compiler inserts s_wait_dscnt

    // P (16x32) as A-fragment from LDS, v (32x16; 8 valid cols) as B-fragment
    v16h pa = make_a_frag(&Plds[w][ln][16 * half], &Plds[w][ln][16 * half + 8]);
    v16h vb = {};
    if (ln < 8) vb = *(const v16h*)(vt + (nh * 8 + ln) * 1024 + t0 + 16 * half);
    acc = __builtin_amdgcn_wmma_f32_16x16x32_f16(false, pa, false, vb, (short)0, acc, false, false);
  }

#pragma unroll
  for (int r = 0; r < 8; ++r)
    if (ln < 8) {
      float v = acc[r] / l_run[r];
      yt[((size_t)n * 1024 + s0 + r + 8 * half) * 256 + head * 8 + ln] = (f16)v;
    }
}

// ---------------- out projection + bias + residual ----------------
__global__ void out_gemm_kernel(const f16* __restrict__ W,      // [256][256]
                                const float* __restrict__ bias, // [256]
                                const f16* __restrict__ yt,     // [n][1024][256]
                                const float* __restrict__ x,    // [n][256][1024]
                                float* __restrict__ out)        // [n][256][1024]
{
  const int n    = blockIdx.z;
  const int lane = threadIdx.x & 31;
  const int w    = threadIdx.x >> 5;
  const int half = lane >> 4;
  const int ln   = lane & 15;
  const int m0   = blockIdx.x * 128 + (w & 1) * 64;
  const int n0   = blockIdx.y * 128 + (w >> 1) * 32;

  v8f acc[4][2] = {};
  const f16* yb = yt + (size_t)n * 1024 * 256;

  for (int kk = 0; kk < 256; kk += 32) {
    v16h bfrag[2];
#pragma unroll
    for (int nt = 0; nt < 2; ++nt)
      bfrag[nt] = *(const v16h*)(yb + (size_t)(n0 + nt * 16 + ln) * 256 + kk + 16 * half);
#pragma unroll
    for (int mt = 0; mt < 4; ++mt) {
      const f16* rowp = W + (size_t)(m0 + mt * 16 + ln) * 256 + kk;
      v16h afrag = make_a_frag(rowp + 16 * half, rowp + 16 * half + 8);
#pragma unroll
      for (int nt = 0; nt < 2; ++nt)
        acc[mt][nt] = __builtin_amdgcn_wmma_f32_16x16x32_f16(
            false, afrag, false, bfrag[nt], (short)0, acc[mt][nt], false, false);
    }
  }

#pragma unroll
  for (int mt = 0; mt < 4; ++mt)
#pragma unroll
    for (int nt = 0; nt < 2; ++nt)
#pragma unroll
      for (int r = 0; r < 8; ++r) {
        int o = m0 + mt * 16 + r + 8 * half;
        int t = n0 + nt * 16 + ln;
        size_t idx = ((size_t)n * 256 + o) * 1024 + t;
        out[idx] = acc[mt][nt][r] + bias[o] + x[idx];
      }
}

extern "C" void kernel_launch(void* const* d_in, const int* in_sizes, int n_in,
                              void* d_out, int out_size, void* d_ws, size_t ws_size,
                              hipStream_t stream) {
  (void)in_sizes; (void)n_in; (void)out_size; (void)ws_size;
  const float* x     = (const float*)d_in[0];
  const float* gn_w  = (const float*)d_in[1];
  const float* gn_b  = (const float*)d_in[2];
  const float* qkv_w = (const float*)d_in[3];
  const float* qkv_b = (const float*)d_in[4];
  const float* out_w = (const float*)d_in[5];
  const float* out_b = (const float*)d_in[6];
  float* out = (float*)d_out;

  // workspace layout (f16): xnt | qkv_w16 | out_w16 | qt | kt | vt ; yt aliases xnt
  f16* xnt    = (f16*)d_ws;                          // 8*1024*256
  f16* qkvw16 = xnt + (size_t)8 * 1024 * 256;        // 768*256
  f16* outw16 = qkvw16 + (size_t)768 * 256;          // 256*256
  f16* qt     = outw16 + (size_t)256 * 256;          // 8*32*1024*8
  f16* kt     = qt + (size_t)8 * 32 * 1024 * 8;
  f16* vt     = kt + (size_t)8 * 32 * 1024 * 8;
  f16* yt     = xnt;  // xnt fully consumed by qkv_gemm before attn writes yt

  cvt_f16_kernel<<<192, 256, 0, stream>>>(qkv_w, qkvw16, 768 * 256);
  cvt_f16_kernel<<<64, 256, 0, stream>>>(out_w, outw16, 256 * 256);
  groupnorm_kernel<<<dim3(8, 8), 256, 0, stream>>>(x, gn_w, gn_b, xnt);
  qkv_gemm_kernel<<<dim3(6, 8, 8), 256, 0, stream>>>(qkvw16, qkv_b, xnt, qt, kt, vt);
  attn_kernel<<<dim3(8, 32, 8), 256, 0, stream>>>(qt, kt, vt, yt);
  out_gemm_kernel<<<dim3(2, 8, 8), 256, 0, stream>>>(outw16, out_b, yt, x, out);
}
